// MultiHeadAttentionV1_24120536334829
// MI455X (gfx1250) — matrix operand order repaired and checked by reading, not compiled, not run on hardware
//
#include <hip/hip_runtime.h>
#include <hip/hip_fp16.h>
#include <stdint.h>

typedef _Float16 f16;
typedef __attribute__((ext_vector_type(16))) _Float16 v16h;
typedef __attribute__((ext_vector_type(8)))  float    v8f;

#define BB   2
#define SS   2048
#define DIN  768
#define HH   12
#define DO   64

// f16-element offsets inside workspace
#define N_XH    (BB*SS*DIN)        // 3,145,728
#define N_W     (HH*DO*DIN)        //   589,824 per matrix (transposed [H,64,768])
#define N_QKV   (BB*HH*SS*DO)      // 3,145,728
#define OFF_XH  0
#define OFF_WQT (OFF_XH + N_XH)
#define OFF_WKT (OFF_WQT + N_W)
#define OFF_WVT (OFF_WKT + N_W)
#define OFF_Q   (OFF_WVT + N_W)
#define OFF_K   (OFF_Q + N_QKV)
#define OFF_VT  (OFF_K + N_QKV)
// total f16 elems = 14,352,384  (~28.7 MB)

union Frag { v16h h; uint32_t u[8]; };
static __device__ __forceinline__ v8f vzero() {
  v8f z = {0.f,0.f,0.f,0.f,0.f,0.f,0.f,0.f}; return z;
}

// A-fragment (16x32 f16): lane row = lane%16; K = 16*(v/4)+2*(v%4)+8*half+p
static __device__ __forceinline__ void loadA(Frag& a, const uint32_t* __restrict__ base,
                                             int kk, int half) {
  #pragma unroll
  for (int v = 0; v < 8; ++v)
    a.u[v] = base[(kk + 16 * (v >> 2) + 2 * (v & 3) + 8 * half) >> 1];
}
// B-fragment (32x16 f16): lane col = lane%16; K = 16*(v/4)+2*(v%4)+8*half+p
static __device__ __forceinline__ void loadB(Frag& b, const uint32_t* __restrict__ base,
                                             int kk, int half) {
  #pragma unroll
  for (int v = 0; v < 8; ++v)
    b.u[v] = base[(kk + 16 * (v >> 2) + 2 * (v & 3) + 8 * half) >> 1];
}
#define WMMA(A, B, C) __builtin_amdgcn_wmma_f32_16x16x32_f16( \
    false, (A).h, false, (B).h, (short)0, (C), false, false)

// ---------------------------------------------------------------- convert ---
__global__ __launch_bounds__(256) void cvt_kernel(
    const float* __restrict__ x, const float* __restrict__ Wq,
    const float* __restrict__ Wk, const float* __restrict__ Wv,
    f16* __restrict__ ws) {
  const int total = N_XH + 3 * N_W;
  for (int i = blockIdx.x * blockDim.x + threadIdx.x; i < total;
       i += gridDim.x * blockDim.x) {
    if (i < N_XH) {
      ws[OFF_XH + i] = (f16)x[i];
    } else {
      int j = i - N_XH;
      int m = j / N_W;                 // 0=Wq 1=Wk 2=Wv
      int o = j % N_W;
      int h = o / (DO * DIN);
      int r = o % (DO * DIN);
      int e = r / DIN;                 // out feature
      int d = r % DIN;                 // in feature
      const float* W = (m == 0) ? Wq : (m == 1) ? Wk : Wv;
      ws[OFF_WQT + j] = (f16)W[(h * DIN + d) * DO + e];   // transpose -> [H,64,768]
    }
  }
}

// ------------------------------------------------------- Q / K projection ---
// wave tiles: mat(2) x head(12) x mtile(256).  Wave: 16 rows x 64 cols out.
// 3-deep buffer rotation: a buffer is rewritten two WMMA-groups after its
// last read, outside the WMMA WAR-hazard window -> no copies / v_nops, and
// loads stay a full K-step ahead of the WMMAs.
__global__ __launch_bounds__(256) void qk_proj_kernel(f16* __restrict__ ws) {
  const int wave = (blockIdx.x * blockDim.x + threadIdx.x) >> 5;
  const int lane = threadIdx.x & 31;
  const int ln = lane & 15, half = lane >> 4;
  const int mt  = wave & 255;
  const int h   = (wave >> 8) % HH;
  const int mat = (wave >> 8) / HH;

  const f16* __restrict__ xh = ws + OFF_XH;
  const f16* __restrict__ Wt = ws + (size_t)OFF_WQT + (size_t)mat * N_W;
  f16* __restrict__ Out      = ws + (mat == 0 ? (size_t)OFF_Q : (size_t)OFF_K);

  v8f acc[4];
  #pragma unroll
  for (int i = 0; i < 4; ++i) acc[i] = vzero();

  const int arow = mt * 16 + ln;                       // flattened B*S row
  const uint32_t* __restrict__ xrow =
      (const uint32_t*)(xh + (size_t)arow * DIN);
  const uint32_t* __restrict__ wrow[4];
  #pragma unroll
  for (int nt = 0; nt < 4; ++nt)
    wrow[nt] = (const uint32_t*)(Wt + (size_t)(h * DO + nt * 16 + ln) * DIN);

  auto loadStep = [&](Frag& a, Frag (&bb)[4], int kk) {
    loadA(a, xrow, kk, half);
    #pragma unroll
    for (int nt = 0; nt < 4; ++nt) loadB(bb[nt], wrow[nt], kk, half);
  };
  auto mmaStep = [&](Frag& a, Frag (&bb)[4]) {
    #pragma unroll
    for (int nt = 0; nt < 4; ++nt) acc[nt] = WMMA(a, bb[nt], acc[nt]);
  };

  Frag aA, aB, aC, bA[4], bB[4], bC[4];
  loadStep(aA, bA, 0);
  loadStep(aB, bB, 32);
  #pragma unroll 1
  for (int kk = 0; kk < DIN - 96; kk += 96) {          // kk = 0..576, 7 iters
    loadStep(aC, bC, kk + 64);  mmaStep(aA, bA);       // step kk
    loadStep(aA, bA, kk + 96);  mmaStep(aB, bB);       // step kk+32
    loadStep(aB, bB, kk + 128); mmaStep(aC, bC);       // step kk+64
  }
  loadStep(aC, bC, DIN - 32);                          // step 736
  mmaStep(aA, bA);                                     // step 672
  mmaStep(aB, bB);                                     // step 704
  mmaStep(aC, bC);                                     // step 736

  #pragma unroll
  for (int nt = 0; nt < 4; ++nt) {
    #pragma unroll
    for (int r = 0; r < 8; ++r) {
      int srow = mt * 16 + r + 8 * half;               // flattened B*S
      int bi = srow / SS, s = srow % SS;
      int e  = nt * 16 + ln;
      Out[(((size_t)(bi * HH + h) * SS + s) * DO) + e] = (f16)acc[nt][r];
    }
  }
}

// ---------------------------------------------------- V projection (V^T) ---
// Vt[b,h,e,s] = sum_d Wv[h,d,e] * x[b,s,d]   (A = Wv^T rows=e, B = x^T cols=s)
// wave tiles: head(12) x etile(4) x sblock(64).  Wave: 16 e x 64 s out.
__global__ __launch_bounds__(256) void v_proj_kernel(f16* __restrict__ ws) {
  const int wave = (blockIdx.x * blockDim.x + threadIdx.x) >> 5;
  const int lane = threadIdx.x & 31;
  const int ln = lane & 15, half = lane >> 4;
  const int sb = wave & 63;
  const int et = (wave >> 6) & 3;
  const int h  = wave >> 8;

  const f16* __restrict__ xh  = ws + OFF_XH;
  const f16* __restrict__ Wvt = ws + OFF_WVT;
  f16* __restrict__ Vt        = ws + OFF_VT;

  v8f acc[4];
  #pragma unroll
  for (int i = 0; i < 4; ++i) acc[i] = vzero();

  const uint32_t* __restrict__ wrow =
      (const uint32_t*)(Wvt + (size_t)(h * DO + et * 16 + ln) * DIN);
  const uint32_t* __restrict__ xrow[4];
  #pragma unroll
  for (int nt = 0; nt < 4; ++nt)
    xrow[nt] = (const uint32_t*)(xh + (size_t)(sb * 64 + nt * 16 + ln) * DIN);

  auto loadStep = [&](Frag& a, Frag (&bb)[4], int kk) {
    loadA(a, wrow, kk, half);
    #pragma unroll
    for (int nt = 0; nt < 4; ++nt) loadB(bb[nt], xrow[nt], kk, half);
  };
  auto mmaStep = [&](Frag& a, Frag (&bb)[4]) {
    #pragma unroll
    for (int nt = 0; nt < 4; ++nt) acc[nt] = WMMA(a, bb[nt], acc[nt]);
  };

  Frag aA, aB, aC, bA[4], bB[4], bC[4];
  loadStep(aA, bA, 0);
  loadStep(aB, bB, 32);
  #pragma unroll 1
  for (int kk = 0; kk < DIN - 96; kk += 96) {
    loadStep(aC, bC, kk + 64);  mmaStep(aA, bA);
    loadStep(aA, bA, kk + 96);  mmaStep(aB, bB);
    loadStep(aB, bB, kk + 128); mmaStep(aC, bC);
  }
  loadStep(aC, bC, DIN - 32);
  mmaStep(aA, bA);
  mmaStep(aB, bB);
  mmaStep(aC, bC);

  #pragma unroll
  for (int nt = 0; nt < 4; ++nt) {
    #pragma unroll
    for (int r = 0; r < 8; ++r) {
      int e  = et * 16 + r + 8 * half;
      int sg = sb * 64 + nt * 16 + ln;                 // flattened B*S
      int bi = sg / SS, s = sg % SS;
      Vt[(((size_t)(bi * HH + h) * DO + e) * SS) + s] = (f16)acc[nt][r];
    }
  }
}

// ---------------------------------------------------------- attention ------
// grid: B*H*(S/64) blocks, 4 waves; each wave owns 16 query rows.
// Pipeline per 32-key block: V frags issued, QK^T WMMAs on preloaded K frags,
// next block's K frags issued (hidden behind softmax VALU), online softmax,
// P re-layout via wave-private LDS, P*V WMMAs.
__global__ __launch_bounds__(128) void attn_kernel(const f16* __restrict__ ws,
                                                   float* __restrict__ out) {
  __shared__ f16 Plds[4][16][32];                      // wave-private 1KB each
  const int w = threadIdx.x >> 5, lane = threadIdx.x & 31;
  const int ln = lane & 15, half = lane >> 4;
  const int mblk = blockIdx.x & 31;
  const int h = (blockIdx.x >> 5) % HH;
  const int b = blockIdx.x / (32 * HH);
  const int q0 = mblk * 64 + w * 16;

  const size_t bh = (size_t)(b * HH + h);
  const f16* __restrict__ Q  = ws + OFF_Q  + bh * SS * DO;
  const f16* __restrict__ K  = ws + OFF_K  + bh * SS * DO;
  const f16* __restrict__ Vt = ws + OFF_VT + bh * DO * SS;

  // Q fragments: features [0,32) and [32,64)
  Frag qa[2];
  const uint32_t* __restrict__ qrow =
      (const uint32_t*)(Q + (size_t)(q0 + ln) * DO);
  loadA(qa[0], qrow, 0, half);
  loadA(qa[1], qrow, 32, half);

  const uint32_t* __restrict__ vrow[4];
  #pragma unroll
  for (int nt = 0; nt < 4; ++nt)
    vrow[nt] = (const uint32_t*)(Vt + (size_t)(nt * 16 + ln) * SS);

  v8f acc[4];
  #pragma unroll
  for (int i = 0; i < 4; ++i) acc[i] = vzero();
  float mrow[8], lrow[8];
  #pragma unroll
  for (int r = 0; r < 8; ++r) { mrow[r] = -__builtin_inff(); lrow[r] = 0.f; }

  const int njb = (q0 + 16 + 31) >> 5;                 // causal: keys < q0+16

  auto loadK = [&](Frag (&kb)[2][2], int k0) {
    #pragma unroll
    for (int t = 0; t < 2; ++t) {
      const uint32_t* __restrict__ krow =
          (const uint32_t*)(K + (size_t)(k0 + t * 16 + ln) * DO);
      loadB(kb[t][0], krow, 0, half);
      loadB(kb[t][1], krow, 32, half);
    }
  };

  auto step = [&](int jb, Frag (&kb)[2][2], Frag (&kbn)[2][2]) {
    const int k0 = jb * 32;
    // ---- V fragments: in flight through QK WMMAs + softmax ----
    Frag vb[4];
    #pragma unroll
    for (int nt = 0; nt < 4; ++nt) loadB(vb[nt], vrow[nt], k0, half);

    // ---- scores: S = Q K^T (K frags preloaded last iteration) ----
    v8f sc[2];
    #pragma unroll
    for (int t = 0; t < 2; ++t) {
      v8f c = vzero();
      c = WMMA(qa[0], kb[t][0], c);
      c = WMMA(qa[1], kb[t][1], c);
      sc[t] = c;
    }
    // ---- issue next block's K frags; latency hidden by softmax below ----
    if (jb + 1 < njb) loadK(kbn, k0 + 32);
    if (jb + 2 < njb)
      __builtin_prefetch(K + (size_t)(k0 + 64 + ln) * DO, 0, 1);

    // ---- online softmax (row = r + 8*half, col = lane%16 per C layout) ----
    #pragma unroll
    for (int r = 0; r < 8; ++r) {
      const int qr = q0 + r + 8 * half;
      float s0 = sc[0][r] * 0.125f;                    // 1/sqrt(64)
      float s1 = sc[1][r] * 0.125f;
      if (k0 + ln      > qr) s0 = -__builtin_inff();
      if (k0 + 16 + ln > qr) s1 = -__builtin_inff();
      float mx = fmaxf(s0, s1);
      #pragma unroll
      for (int d = 1; d < 16; d <<= 1) mx = fmaxf(mx, __shfl_xor(mx, d, 32));
      const float mnew = fmaxf(mrow[r], mx);
      const float corr = __expf(mrow[r] - mnew);
      const float p0 = __expf(s0 - mnew);
      const float p1 = __expf(s1 - mnew);
      float rs = p0 + p1;
      #pragma unroll
      for (int d = 1; d < 16; d <<= 1) rs += __shfl_xor(rs, d, 32);
      lrow[r] = lrow[r] * corr + rs;
      mrow[r] = mnew;
      #pragma unroll
      for (int nt = 0; nt < 4; ++nt) acc[nt][r] *= corr;
      Plds[w][r + 8 * half][ln]      = (f16)p0;        // re-layout via LDS
      Plds[w][r + 8 * half][16 + ln] = (f16)p1;
    }
    asm volatile("" ::: "memory");  // keep DS loads after DS stores (HW in-order per wave)

    // ---- P (A-frag from LDS) x V^T ----
    Frag pa;
    #pragma unroll
    for (int v = 0; v < 8; ++v)
      pa.u[v] = *(const uint32_t*)&Plds[w][ln]
                    [16 * (v >> 2) + 2 * (v & 3) + 8 * half];
    #pragma unroll
    for (int nt = 0; nt < 4; ++nt) acc[nt] = WMMA(pa, vb[nt], acc[nt]);
  };

  Frag kbu[2][2], kbv[2][2];
  loadK(kbu, 0);
  #pragma unroll 1
  for (int jb = 0; jb < njb; jb += 2) {                // explicit ping-pong
    step(jb, kbu, kbv);
    if (jb + 1 < njb) step(jb + 1, kbv, kbu);
  }

  // ---- normalize + store out[b, s, h*64+e] (f32) ----
  #pragma unroll
  for (int r = 0; r < 8; ++r) {
    const float inv = 1.f / lrow[r];
    const int srow = q0 + r + 8 * half;
    #pragma unroll
    for (int nt = 0; nt < 4; ++nt) {
      out[((size_t)(b * SS + srow) * (HH * DO)) + h * DO + nt * 16 + ln] =
          acc[nt][r] * inv;
    }
  }
}

// ---------------------------------------------------------------------------
extern "C" void kernel_launch(void* const* d_in, const int* in_sizes, int n_in,
                              void* d_out, int out_size, void* d_ws, size_t ws_size,
                              hipStream_t stream) {
  const float* x  = (const float*)d_in[0];
  const float* Wq = (const float*)d_in[1];
  const float* Wk = (const float*)d_in[2];
  const float* Wv = (const float*)d_in[3];
  float* out = (float*)d_out;
  f16* ws = (f16*)d_ws;

  cvt_kernel<<<2048, 256, 0, stream>>>(x, Wq, Wk, Wv, ws);
  qk_proj_kernel<<<768, 256, 0, stream>>>(ws);   // 6144 waves = 2*12*256 tiles
  v_proj_kernel<<<384, 256, 0, stream>>>(ws);    // 3072 waves = 12*4*64 tiles
  attn_kernel<<<BB * HH * (SS / 64), 128, 0, stream>>>(ws, out);
  (void)in_sizes; (void)n_in; (void)out_size; (void)ws_size;
}
